// PrototypicalHead_3770981286406
// MI455X (gfx1250) — matrix operand order repaired and checked by reading, not compiled, not run on hardware
//
#include <hip/hip_runtime.h>
#include <hip/hip_bf16.h>
#include <math.h>

// ---------- CDNA5 WMMA vector types ----------
typedef __attribute__((ext_vector_type(16))) __bf16 v16bf;
typedef __attribute__((ext_vector_type(8)))  float  v8f;

#define NQ    65536
#define NWAY  5
#define NKPT  17
#define DIM   512
#define NSUP  25

// output layout (flat concat, reference return order)
constexpr size_t OFF_CONF  = (size_t)NQ * NKPT * 2;            // 2228224
constexpr size_t OFF_DIST  = OFF_CONF + (size_t)NQ * NKPT;     // 3342336
constexpr size_t OFF_CLS   = OFF_DIST + (size_t)NQ * NWAY * NKPT; // 8912896
constexpr size_t OFF_PROTO = OFF_CLS + (size_t)NQ;             // 8978432

__device__ __forceinline__ unsigned short f2bf(float f) {
  unsigned u = __float_as_uint(f);
  u += 0x7FFFu + ((u >> 16) & 1u);   // round-to-nearest-even
  return (unsigned short)(u >> 16);
}
__device__ __forceinline__ float bf2f(unsigned short h) {
  return __uint_as_float(((unsigned)h) << 16);
}
__device__ __forceinline__ float sigmoidf(float x) {
  return 1.f / (1.f + __expf(-x));
}

// ---------------------------------------------------------------------------
// Kernel 1: pack [Wo1 | Wc1] (512x256 each, K-major) into N-major bf16 Wt[512][512]
// so B-fragments are contiguous per lane.
// ---------------------------------------------------------------------------
__global__ void prep_weights(const float* __restrict__ Wo1,
                             const float* __restrict__ Wc1,
                             unsigned short* __restrict__ Wt) {
  int idx = blockIdx.x * blockDim.x + threadIdx.x;  // 0 .. 512*512-1
  int n = idx >> 9;       // output column (fan_out)
  int k = idx & 511;      // input dim (fan_in)
  float v = (n < 256) ? Wo1[(size_t)k * 256 + n] : Wc1[(size_t)k * 256 + (n - 256)];
  Wt[(size_t)n * DIM + k] = f2bf(v);
}

// ---------------------------------------------------------------------------
// Kernel 2: prototype net (tiny): h = relu(sf@W1+b1)@W2+b2, mean per way,
// write prototypes output + proto vectors + norms to workspace.
// ---------------------------------------------------------------------------
__global__ void proto_kernel(const float* __restrict__ sf,
                             const float* __restrict__ W1, const float* __restrict__ b1,
                             const float* __restrict__ W2, const float* __restrict__ b2,
                             float* __restrict__ protoW, float* __restrict__ pn,
                             float* __restrict__ protoOut) {
  __shared__ float h1[NSUP * DIM];
  __shared__ float h2[NSUP * DIM];
  const int t = threadIdx.x;

  for (int i = t; i < NSUP * DIM; i += 256) {
    int r = i >> 9, c = i & 511;
    float acc = b1[c];
    for (int k = 0; k < DIM; ++k) acc += sf[(size_t)r * DIM + k] * W1[(size_t)k * DIM + c];
    h1[i] = acc > 0.f ? acc : 0.f;
  }
  __syncthreads();
  for (int i = t; i < NSUP * DIM; i += 256) {
    int r = i >> 9, c = i & 511;
    float acc = b2[c];
    for (int k = 0; k < DIM; ++k) acc += h1[r * DIM + k] * W2[(size_t)k * DIM + c];
    h2[i] = acc;
  }
  __syncthreads();
  for (int i = t; i < NWAY * DIM; i += 256) {
    int w = i >> 9, d = i & 511;
    float s = 0.f;
    for (int sh = 0; sh < 5; ++sh) s += h2[(w * 5 + sh) * DIM + d];
    float pv = s * 0.2f;
    protoW[i] = pv;
    for (int k = 0; k < NKPT; ++k) protoOut[((size_t)(w * NKPT + k)) * DIM + d] = pv;
  }
  __syncthreads();
  if (t < NWAY) {
    float s = 0.f;
    for (int d = 0; d < DIM; ++d) { float v = protoW[t * DIM + d]; s += v * v; }
    pn[t] = sqrtf(s);
  }
}

// ---------------------------------------------------------------------------
// Kernel 3: main query kernel. 256 threads = 8 wave32s, 16 rows per wave,
// 128 query rows per workgroup. Layer1 GEMM in bf16 WMMA, epilogue in LDS,
// then per-row second layers + cosine distances + outputs.
// ---------------------------------------------------------------------------
__global__ __launch_bounds__(256) void query_kernel(
    const float* __restrict__ q, const float* __restrict__ ic,
    const unsigned short* __restrict__ Wt,
    const float* __restrict__ protoW, const float* __restrict__ pn,
    const float* __restrict__ bo1, const float* __restrict__ bc1,
    const float* __restrict__ Wo2, const float* __restrict__ bo2,
    const float* __restrict__ Wc2, const float* __restrict__ bc2,
    const float* __restrict__ temperature,
    float* __restrict__ out) {
  __shared__ unsigned short hLds[128][520];  // relu(H+bias) in bf16, padded rows
  __shared__ float biasS[512];               // [bo1 | bc1]

  const int t = threadIdx.x;
  const int lane = t & 31;
  const int wave = t >> 5;

  for (int i = t; i < 512; i += 256) biasS[i] = (i < 256) ? bo1[i] : bc1[i - 256];
  __syncthreads();

  const int rowBase = blockIdx.x * 128 + wave * 16;
  const int mrow    = rowBase + (lane & 15);
  const int hiHalf  = (lane & 16);           // 0 for lanes 0-15, 16 for 16-31

  // ---- A fragments: 16x32 bf16 per K-step, 16 steps resident in VGPRs ----
  union AF { v16bf v; unsigned short h[16]; };
  v16bf aF[16];
  {
    const float* qr = q + (size_t)mrow * DIM + hiHalf;  // +0 or +16
    #pragma unroll
    for (int s = 0; s < 16; ++s) {
      AF a;
      const float* p = qr + s * 32;
      #pragma unroll
      for (int j = 0; j < 8; ++j) a.h[j]     = f2bf(p[j]);       // K base..base+7
      #pragma unroll
      for (int j = 0; j < 8; ++j) a.h[8 + j] = f2bf(p[8 + j]);  // K base+8..+15
      aF[s] = a.v;
    }
  }

  union BF { v16bf v; uint4 u[2]; };
  const int nLane = lane & 15;
  const int kB    = hiHalf;  // +0 / +16 inside each 32-K step

  #pragma unroll 1
  for (int nc = 0; nc < 8; ++nc) {
    const int nBase = nc * 64;
    if (nc < 7)
      __builtin_prefetch(Wt + (size_t)(nBase + 64 + nLane) * DIM, 0, 1);

    v8f acc0 = {}, acc1 = {}, acc2 = {}, acc3 = {};
    const unsigned short* w0 = Wt + (size_t)(nBase + nLane) * DIM + kB;

    #pragma unroll
    for (int s = 0; s < 16; ++s) {
      BF fb0, fb1, fb2, fb3;
      const uint4* p0 = (const uint4*)(w0 + s * 32);
      const uint4* p1 = (const uint4*)(w0 + 16 * DIM + s * 32);
      const uint4* p2 = (const uint4*)(w0 + 32 * DIM + s * 32);
      const uint4* p3 = (const uint4*)(w0 + 48 * DIM + s * 32);
      fb0.u[0] = p0[0]; fb0.u[1] = p0[1];
      fb1.u[0] = p1[0]; fb1.u[1] = p1[1];
      fb2.u[0] = p2[0]; fb2.u[1] = p2[1];
      fb3.u[0] = p3[0]; fb3.u[1] = p3[1];
      acc0 = __builtin_amdgcn_wmma_f32_16x16x32_bf16(false, aF[s], false, fb0.v, (short)0, acc0, false, false);
      acc1 = __builtin_amdgcn_wmma_f32_16x16x32_bf16(false, aF[s], false, fb1.v, (short)0, acc1, false, false);
      acc2 = __builtin_amdgcn_wmma_f32_16x16x32_bf16(false, aF[s], false, fb2.v, (short)0, acc2, false, false);
      acc3 = __builtin_amdgcn_wmma_f32_16x16x32_bf16(false, aF[s], false, fb3.v, (short)0, acc3, false, false);
    }

    // epilogue: bias + relu -> LDS bf16. C layout: vgpr r, lanes 0-15 => M=r, 16-31 => M=r+8
    const int mLoc = wave * 16 + (hiHalf >> 1);
    const int n0 = nBase + nLane;
    #pragma unroll
    for (int r = 0; r < 8; ++r) {
      int rr = mLoc + r;
      float v0 = acc0[r] + biasS[n0];
      float v1 = acc1[r] + biasS[n0 + 16];
      float v2 = acc2[r] + biasS[n0 + 32];
      float v3 = acc3[r] + biasS[n0 + 48];
      hLds[rr][n0]      = f2bf(v0 > 0.f ? v0 : 0.f);
      hLds[rr][n0 + 16] = f2bf(v1 > 0.f ? v1 : 0.f);
      hLds[rr][n0 + 32] = f2bf(v2 > 0.f ? v2 : 0.f);
      hLds[rr][n0 + 48] = f2bf(v3 > 0.f ? v3 : 0.f);
    }
  }
  __syncthreads();

  // ---- phase 2: second layers, cosine distances, outputs (1 thread / row) ----
  if (t < 128) {
    const int grow = blockIdx.x * 128 + t;

    float off0 = bo2[0], off1 = bo2[1], cl = bc2[0];
    for (int n = 0; n < 256; ++n) {
      float h1v = bf2f(hLds[t][n]);
      float h2v = bf2f(hLds[t][256 + n]);
      off0 += h1v * Wo2[n * 2 + 0];
      off1 += h1v * Wo2[n * 2 + 1];
      cl   += h2v * Wc2[n];
    }
    float conf = sigmoidf(cl);

    float qn2 = 0.f;
    float dt[NWAY] = {0.f, 0.f, 0.f, 0.f, 0.f};
    const float* qr = q + (size_t)grow * DIM;
    for (int d = 0; d < DIM; ++d) {
      float qv = qr[d];
      qn2 += qv * qv;
      #pragma unroll
      for (int w = 0; w < NWAY; ++w) dt[w] += qv * protoW[w * DIM + d];
    }
    float qn = sqrtf(qn2);
    float invT = 1.f / temperature[0];

    float dw[NWAY];
    int best = 0; float bestv = 3.4e38f;
    #pragma unroll
    for (int w = 0; w < NWAY; ++w) {
      float den = qn * pn[w];
      den = den > 1e-8f ? den : 1e-8f;
      float sim = dt[w] / den;
      dw[w] = (1.f - sim) * invT;
      if (dw[w] < bestv) { bestv = dw[w]; best = w; }
    }

    float kx = sigmoidf(off0) * ic[(size_t)grow * 2 + 0];
    float ky = sigmoidf(off1) * ic[(size_t)grow * 2 + 1];

    float* kp = out + (size_t)grow * NKPT * 2;
    float* cf = out + OFF_CONF + (size_t)grow * NKPT;
    float* ds = out + OFF_DIST + (size_t)grow * NWAY * NKPT;
    #pragma unroll
    for (int k = 0; k < NKPT; ++k) { kp[k * 2] = kx; kp[k * 2 + 1] = ky; cf[k] = conf; }
    #pragma unroll
    for (int w = 0; w < NWAY; ++w)
      for (int k = 0; k < NKPT; ++k) ds[w * NKPT + k] = dw[w];
    out[OFF_CLS + grow] = (float)best;
  }
}

// ---------------------------------------------------------------------------
extern "C" void kernel_launch(void* const* d_in, const int* in_sizes, int n_in,
                              void* d_out, int out_size, void* d_ws, size_t ws_size,
                              hipStream_t stream) {
  const float* sf   = (const float*)d_in[0];
  // d_in[1] = support_keypoints (unused by reference math)
  const float* q    = (const float*)d_in[2];
  const float* ic   = (const float*)d_in[3];
  const float* W1   = (const float*)d_in[4];
  const float* b1   = (const float*)d_in[5];
  const float* W2   = (const float*)d_in[6];
  const float* b2   = (const float*)d_in[7];
  const float* Wo1  = (const float*)d_in[8];
  const float* bo1  = (const float*)d_in[9];
  const float* Wo2  = (const float*)d_in[10];
  const float* bo2  = (const float*)d_in[11];
  const float* Wc1  = (const float*)d_in[12];
  const float* bc1  = (const float*)d_in[13];
  const float* Wc2  = (const float*)d_in[14];
  const float* bc2  = (const float*)d_in[15];
  const float* temp = (const float*)d_in[16];

  unsigned short* Wt = (unsigned short*)d_ws;                       // 512*512*2 B
  float* protoW = (float*)((char*)d_ws + (size_t)DIM * DIM * 2);    // 5*512 f32
  float* pn     = protoW + NWAY * DIM;                              // 5 f32
  float* out    = (float*)d_out;

  prep_weights<<<dim3((DIM * DIM) / 256), dim3(256), 0, stream>>>(Wo1, Wc1, Wt);
  proto_kernel<<<dim3(1), dim3(256), 0, stream>>>(sf, W1, b1, W2, b2,
                                                  protoW, pn, out + OFF_PROTO);
  query_kernel<<<dim3(NQ / 128), dim3(256), 0, stream>>>(
      q, ic, Wt, protoW, pn, bo1, bc1, Wo2, bo2, Wc2, bc2, temp, out);
}
